// HOGNNEncoder_38225208934609
// MI455X (gfx1250) — matrix operand order repaired and checked by reading, not compiled, not run on hardware
//
#include <hip/hip_runtime.h>
#include <hip/hip_bf16.h>
#include <cstdint>
#include <cstddef>

// ---------------- problem constants ----------------
#define BB   32
#define NNN  64
#define DDD  128
#define EEE  16384
#define LLL  6
#define ROWS (BB * NNN * NNN)            // 131072  (b, i, j) rows
#define XEL  ((size_t)ROWS * DDD)        // 16,777,216 elements of X

typedef __bf16 bf16_t;
typedef __attribute__((ext_vector_type(16))) __bf16 v16bf;
typedef __attribute__((ext_vector_type(8)))  __bf16 v8bf;
typedef __attribute__((ext_vector_type(8)))  float  v8f;

__device__ __forceinline__ bf16_t f2b(float x) { return (bf16_t)x; }

// ---------------- WMMA fragment loaders ----------------
// A fragment (16x32), contiguous k order chosen here: lanes 0-15 -> M=lane, K=0..15;
// lanes 16-31 -> M=lane-16, K=16..31. Two contiguous 8-elem chunks.
__device__ __forceinline__ v16bf load_afrag(const bf16_t* A, int m0, int K, int k0, int lane) {
  const bf16_t* p = A + (size_t)(m0 + (lane & 15)) * K + (k0 + ((lane >> 4) << 4));
  v8bf lo = *(const v8bf*)p;
  v8bf hi = *(const v8bf*)(p + 8);
  v16bf a;
#pragma unroll
  for (int i = 0; i < 8; ++i) { a[i] = lo[i]; a[8 + i] = hi[i]; }
  return a;
}
// B fragment (32x16), same k order as A: lanes 0-15 -> N=lane, K=k0..k0+15 (contiguous);
// lanes 16-31 -> N=lane-16, K=k0+16..k0+31.  Wt is N x K (pre-transposed).
__device__ __forceinline__ v16bf load_bfrag(const bf16_t* Wt, int n0, int K, int k0, int lane) {
  return *(const v16bf*)(Wt + (size_t)(n0 + (lane & 15)) * K + (k0 + ((lane >> 4) << 4)));
}

// ---------------- generic GEMM: C = act(A @ W + bias) ----------------
// A: M x K bf16 row-major. Wt: N x K bf16 (W transposed). ACT 0=none 1=relu 2=silu.
// Each wave computes a 16 x (16*NT) strip: every A fragment is loaded once per
// K-step and reused across NT B-tiles (8x less A traffic than tile-per-wave).
template <int ACT, int OUT_F32, int OUT_BF16, int NT>
__global__ __launch_bounds__(256) void gemm_wmma(const bf16_t* __restrict__ A,
                                                 const bf16_t* __restrict__ Wt,
                                                 const float* __restrict__ bias,
                                                 float* __restrict__ Cf,
                                                 bf16_t* __restrict__ Cb,
                                                 int M, int K, int N) {
  const int lane = threadIdx.x & 31;
  const int wave = threadIdx.x >> 5;
  const int m0 = (blockIdx.y * 8 + wave) * 16;
  const int nbase = blockIdx.x * (16 * NT);
  if (m0 >= M) return;                       // wave-uniform: EXEC stays all-ones
  v8f acc[NT];
#pragma unroll
  for (int t = 0; t < NT; ++t) acc[t] = (v8f){};
  for (int k0 = 0; k0 < K; k0 += 32) {
    __builtin_prefetch(A + (size_t)(m0 + (lane & 15)) * K + k0 + 32, 0, 0);
    v16bf a = load_afrag(A, m0, K, k0, lane);
#pragma unroll
    for (int t = 0; t < NT; ++t) {
      v16bf b = load_bfrag(Wt, nbase + 16 * t, K, k0, lane);
      acc[t] = __builtin_amdgcn_wmma_f32_16x16x32_bf16(false, a, false, b, (short)0, acc[t],
                                                       false, false);
    }
  }
#pragma unroll
  for (int t = 0; t < NT; ++t) {
    const int n = nbase + 16 * t + (lane & 15);
    const float bv = bias ? bias[n] : 0.f;
#pragma unroll
    for (int v = 0; v < 8; ++v) {
      const int m = m0 + v + ((lane >> 4) << 3);
      float x = acc[t][v] + bv;
      if (ACT == 1) x = fmaxf(x, 0.f);
      else if (ACT == 2) x = x / (1.f + __expf(-x));
      if (OUT_F32)  Cf[(size_t)m * N + n] = x;
      if (OUT_BF16) Cb[(size_t)m * N + n] = f2b(x);
    }
  }
}

// ---------------- einsum: per (b,d) 64x64x64 matmuls via WMMA ----------------
// X1 = Am @ Xm  -> Y[:, D..2D) ; X2 = Xm @ Am -> Y[:, 2D..3D)
__global__ __launch_bounds__(256) void einsum_wmma(const bf16_t* __restrict__ Am,
                                                   const bf16_t* __restrict__ AmT,
                                                   const bf16_t* __restrict__ Xm,
                                                   const bf16_t* __restrict__ XmT,
                                                   bf16_t* __restrict__ Y) {
  const int bd = blockIdx.x;                 // b*128 + d, 4096 blocks
  const int b = bd >> 7;
  const int d = bd & 127;
  const int lane = threadIdx.x & 31;
  const int wave = threadIdx.x >> 5;
  const int prod = wave >> 2;                // 0: X1, 1: X2
  const int mi = (wave & 3) * 16;
  const size_t mat = (size_t)bd * 4096;
  const bf16_t* Ap = prod ? (Xm + mat) : (Am + mat);
  const bf16_t* Bp = prod ? (AmT + mat) : (XmT + mat);
  const int colbase = (prod ? 2 * DDD : DDD) + d;
  // A fragments (K=64 -> 2 frags) stay resident across the nj loop.
  v16bf a0 = load_afrag(Ap, mi, 64, 0, lane);
  v16bf a1 = load_afrag(Ap, mi, 64, 32, lane);
  for (int nj = 0; nj < 64; nj += 16) {
    v8f c = {};
    v16bf b0 = load_bfrag(Bp, nj, 64, 0, lane);
    v16bf b1 = load_bfrag(Bp, nj, 64, 32, lane);
    c = __builtin_amdgcn_wmma_f32_16x16x32_bf16(false, a0, false, b0, (short)0, c, false, false);
    c = __builtin_amdgcn_wmma_f32_16x16x32_bf16(false, a1, false, b1, (short)0, c, false, false);
    const int j = nj + (lane & 15);
#pragma unroll
    for (int v = 0; v < 8; ++v) {
      const int i = mi + v + ((lane >> 4) << 3);
      const size_t row = (size_t)b * 4096 + (size_t)i * 64 + j;
      Y[row * (3 * DDD) + colbase] = f2b(c[v]);
    }
  }
}

// ---------------- LayerNorm(+ReLU) over 128-dim rows; wave per row ----------------
template <int RESIDUAL>
__global__ __launch_bounds__(256) void ln_relu(const float* __restrict__ T,
                                               const float* __restrict__ g,
                                               const float* __restrict__ bt,
                                               bf16_t* __restrict__ outb,
                                               float* __restrict__ Xio, int Mrows) {
  const int row = blockIdx.x * 8 + (threadIdx.x >> 5);
  const int lane = threadIdx.x & 31;
  if (row >= Mrows) return;
  const float* t = T + (size_t)row * DDD;
  float v0[4];
  float s = 0.f;
#pragma unroll
  for (int i = 0; i < 4; ++i) { v0[i] = t[lane + 32 * i]; s += v0[i]; }
#pragma unroll
  for (int off = 16; off; off >>= 1) s += __shfl_xor(s, off, 32);
  const float mean = s * (1.f / 128.f);
  float vs = 0.f;
#pragma unroll
  for (int i = 0; i < 4; ++i) { const float dv = v0[i] - mean; vs += dv * dv; }
#pragma unroll
  for (int off = 16; off; off >>= 1) vs += __shfl_xor(vs, off, 32);
  const float inv = rsqrtf(vs * (1.f / 128.f) + 1e-5f);
#pragma unroll
  for (int i = 0; i < 4; ++i) {
    const int d = lane + 32 * i;
    float y = (v0[i] - mean) * inv * g[d] + bt[d];
    y = fmaxf(y, 0.f);
    if (RESIDUAL) Xio[(size_t)row * DDD + d] += y;
    else          outb[(size_t)row * DDD + d] = f2b(y);
  }
}

// ---------------- small helper kernels ----------------
__global__ void fzero(float* p, size_t n) {
  size_t i = (size_t)blockIdx.x * blockDim.x + threadIdx.x;
  if (i < n) p[i] = 0.f;
}

// fp32 (K x N) -> bf16 (N x K)
__global__ void wtrans(const float* __restrict__ W, bf16_t* __restrict__ Wt, int K, int N) {
  int i = blockIdx.x * blockDim.x + threadIdx.x;
  if (i >= K * N) return;
  int n = i % N, k = i / N;
  Wt[(size_t)n * K + k] = f2b(W[i]);
}

__global__ void temb_kernel(const float* t, const float* W1, const float* b1,
                            const float* W2, const float* b2, float* tt) {
  __shared__ float s[DDD];
  const int b = blockIdx.x, d = threadIdx.x;     // 32 blocks x 128 threads
  float h = t[b] * W1[d] + b1[d];
  h = h / (1.f + __expf(-h));
  s[d] = h;
  __syncthreads();
  float acc = b2[d];
  for (int k = 0; k < DDD; ++k) acc += s[k] * W2[k * DDD + d];
  tt[b * DDD + d] = acc / (1.f + __expf(-acc));
}

__global__ void gather_emb(const int* __restrict__ idx, const float* __restrict__ emb,
                           bf16_t* __restrict__ out, size_t n) {
  size_t i = (size_t)blockIdx.x * blockDim.x + threadIdx.x;
  if (i >= n) return;
  int d = (int)(i & 127);
  int e = (int)(i >> 7);
  out[i] = f2b(emb[(size_t)idx[e] * DDD + d]);
}

__global__ void scatter_edges(const int* __restrict__ eidx, const float* __restrict__ Ebuf,
                              float* __restrict__ X) {
  size_t i = (size_t)blockIdx.x * blockDim.x + threadIdx.x;
  if (i >= (size_t)EEE * DDD) return;
  int d = (int)(i & 127);
  int e = (int)(i >> 7);
  int srow = eidx[e];
  int drow = eidx[EEE + e];
  int b = srow / NNN, u = srow % NNN, v = drow % NNN;
  atomicAdd(&X[(((size_t)b * 4096) + (size_t)u * 64 + v) * DDD + d], Ebuf[i]);
}

__global__ void scale_time(float* __restrict__ X, const float* __restrict__ tt) {
  size_t i = (size_t)blockIdx.x * blockDim.x + threadIdx.x;
  if (i >= XEL) return;
  int d = (int)(i & 127);
  int b = (int)(i >> 19);                      // 64*64*128 = 2^19
  X[i] *= tt[b * DDD + d];
}

// X (b,i,j,d) fp32 -> Xm (b,d,i,j) bf16, XmT (b,d,j,i) bf16, optional Y[:,0:D] bf16
__global__ void make_xm(const float* __restrict__ X, bf16_t* __restrict__ Xm,
                        bf16_t* __restrict__ XmT, bf16_t* __restrict__ Y, int writeY) {
  size_t i = (size_t)blockIdx.x * blockDim.x + threadIdx.x;
  if (i >= XEL) return;
  int d = (int)(i & 127);
  size_t r = i >> 7;                 // b*4096 + ii*64 + j
  int j = (int)(r & 63);
  size_t r2 = r >> 6;
  int ii = (int)(r2 & 63);
  int b = (int)(r2 >> 6);
  bf16_t v = f2b(X[i]);
  size_t bd = ((size_t)b * DDD + d) * 4096;
  Xm[bd + (size_t)ii * 64 + j] = v;
  XmT[bd + (size_t)j * 64 + ii] = v;
  if (writeY) Y[r * (3 * DDD) + d] = v;
}

// XsB = bf16(X + X^T over (i,j))
__global__ void symmetrize(const float* __restrict__ X, bf16_t* __restrict__ XsB) {
  size_t i = (size_t)blockIdx.x * blockDim.x + threadIdx.x;
  if (i >= XEL) return;
  int d = (int)(i & 127);
  size_t r = i >> 7;
  int j = (int)(r & 63);
  size_t r2 = r >> 6;
  int ii = (int)(r2 & 63);
  size_t b = r2 >> 6;
  float xt = X[((b * 4096) + (size_t)j * 64 + ii) * DDD + d];
  XsB[i] = f2b(X[i] + xt);
}

__global__ void gather_eout(const int* __restrict__ eidx, const bf16_t* __restrict__ XsB,
                            bf16_t* __restrict__ EoutB) {
  size_t i = (size_t)blockIdx.x * blockDim.x + threadIdx.x;
  if (i >= (size_t)EEE * DDD) return;
  int d = (int)(i & 127);
  int e = (int)(i >> 7);
  int srow = eidx[e];
  int drow = eidx[EEE + e];
  int b = srow / NNN, u = srow % NNN, v = drow % NNN;
  EoutB[i] = XsB[(((size_t)b * 4096) + (size_t)u * 64 + v) * DDD + d];
}

// mean over 64 consecutive rows -> bf16 (2048 x 128)
__global__ void mean_rows64(const float* __restrict__ M2, bf16_t* __restrict__ h2b) {
  const int bi = blockIdx.x, d = threadIdx.x;  // 2048 blocks x 128 threads
  const float* p = M2 + (size_t)bi * 64 * DDD + d;
  float s = 0.f;
  for (int j = 0; j < 64; ++j) s += p[(size_t)j * DDD];
  h2b[(size_t)bi * DDD + d] = f2b(s * (1.f / 64.f));
}

// graph_attr = mean over 64 nodes of node_out
__global__ void graph_mean(const float* __restrict__ node_out, float* __restrict__ gout) {
  const int b = blockIdx.x, d = threadIdx.x;   // 32 blocks x 128 threads
  const float* p = node_out + (size_t)b * 64 * DDD + d;
  float s = 0.f;
  for (int i = 0; i < 64; ++i) s += p[(size_t)i * DDD];
  gout[(size_t)b * DDD + d] = s * (1.f / 64.f);
}

// ---------------- host orchestration ----------------
static inline unsigned gblk(size_t n) { return (unsigned)((n + 255) / 256); }

extern "C" void kernel_launch(void* const* d_in, const int* in_sizes, int n_in,
                              void* d_out, int out_size, void* d_ws, size_t ws_size,
                              hipStream_t stream) {
  (void)in_sizes; (void)n_in; (void)out_size; (void)ws_size;
  const int*   edge_index = (const int*)  d_in[1];
  const int*   edge_attr  = (const int*)  d_in[2];
  const float* t          = (const float*)d_in[3];
  const float* edge_emb   = (const float*)d_in[5];
  const float* temb_W1 = (const float*)d_in[6];
  const float* temb_b1 = (const float*)d_in[7];
  const float* temb_W2 = (const float*)d_in[8];
  const float* temb_b2 = (const float*)d_in[9];
  // nin_* (10..13), node_emb (4), x (0) are dead code in the reference.
  const float* ein_W1  = (const float*)d_in[14];
  const float* ein_b1  = (const float*)d_in[15];
  const float* ein_W2  = (const float*)d_in[16];
  const float* ein_b2  = (const float*)d_in[17];
  const float* conv_W1 = (const float*)d_in[18];
  const float* conv_b1 = (const float*)d_in[19];
  const float* conv_g1 = (const float*)d_in[20];
  const float* conv_bt1= (const float*)d_in[21];
  const float* conv_W2 = (const float*)d_in[22];
  const float* conv_b2 = (const float*)d_in[23];
  const float* conv_g2 = (const float*)d_in[24];
  const float* conv_bt2= (const float*)d_in[25];
  const float* mlp_W1  = (const float*)d_in[26];
  const float* mlp_b1  = (const float*)d_in[27];
  const float* mlp_W2  = (const float*)d_in[28];
  const float* mlp_b2  = (const float*)d_in[29];
  const float* fn_W1   = (const float*)d_in[30];
  const float* fn_b1   = (const float*)d_in[31];
  const float* fn_W2   = (const float*)d_in[32];
  const float* fn_b2   = (const float*)d_in[33];

  char* ws = (char*)d_ws;
  size_t off = 0;
  auto alloc = [&](size_t bytes) -> char* {
    char* p = ws + off;
    off += (bytes + 255) & ~(size_t)255;
    return p;
  };

  float*  tt      = (float*) alloc((size_t)BB * DDD * 4);
  bf16_t* wEin1t  = (bf16_t*)alloc((size_t)256 * 128 * 2);
  bf16_t* wEin2t  = (bf16_t*)alloc((size_t)128 * 256 * 2);
  bf16_t* wConv1t = (bf16_t*)alloc((size_t)LLL * 128 * 384 * 2);
  bf16_t* wConv2t = (bf16_t*)alloc((size_t)LLL * 128 * 128 * 2);
  bf16_t* wMlp1t  = (bf16_t*)alloc((size_t)128 * 128 * 2);
  bf16_t* wMlp2t  = (bf16_t*)alloc((size_t)128 * 128 * 2);
  bf16_t* wFn1t   = (bf16_t*)alloc((size_t)256 * 128 * 2);
  bf16_t* wFn2t   = (bf16_t*)alloc((size_t)128 * 256 * 2);
  bf16_t* E0      = (bf16_t*)alloc((size_t)EEE * 128 * 2);
  bf16_t* Etmp    = (bf16_t*)alloc((size_t)EEE * 256 * 2);
  float*  Ebuf    = (float*) alloc((size_t)EEE * 128 * 4);
  float*  X       = (float*) alloc(XEL * 4);
  bf16_t* Am      = (bf16_t*)alloc(XEL * 2);
  bf16_t* AmT     = (bf16_t*)alloc(XEL * 2);
  bf16_t* Xm      = (bf16_t*)alloc(XEL * 2);
  bf16_t* XmT     = (bf16_t*)alloc(XEL * 2);
  bf16_t* Y       = (bf16_t*)alloc((size_t)ROWS * 384 * 2);
  float*  T1      = (float*) alloc(XEL * 4);
  bf16_t* T1b     = (bf16_t*)alloc(XEL * 2);
  float*  T2      = (float*) alloc(XEL * 4);
  bf16_t* XsB     = (bf16_t*)alloc(XEL * 2);
  bf16_t* EoutB   = (bf16_t*)alloc((size_t)EEE * 128 * 2);
  bf16_t* h2b     = (bf16_t*)alloc((size_t)2048 * 128 * 2);
  bf16_t* F1      = (bf16_t*)alloc((size_t)EEE * 256 * 2);
  bf16_t* M1b = T1b;      // reuse after layer loop
  float*  M2  = T1;       // reuse after layer loop

  float* node_out = (float*)d_out;                       // 2048*128
  float* edge_out = node_out + (size_t)2048 * 128;       // 16384*128
  float* gattr    = edge_out + (size_t)EEE * 128;        // 32*128

  const dim3 blk(256);

  // -- weight transposes to bf16 N x K --
  wtrans<<<gblk(128 * 256), blk, 0, stream>>>(ein_W1, wEin1t, 128, 256);
  wtrans<<<gblk(256 * 128), blk, 0, stream>>>(ein_W2, wEin2t, 256, 128);
  for (int l = 0; l < LLL; ++l) {
    wtrans<<<gblk(384 * 128), blk, 0, stream>>>(conv_W1 + (size_t)l * 384 * 128,
                                                wConv1t + (size_t)l * 128 * 384, 384, 128);
    wtrans<<<gblk(128 * 128), blk, 0, stream>>>(conv_W2 + (size_t)l * 128 * 128,
                                                wConv2t + (size_t)l * 128 * 128, 128, 128);
  }
  wtrans<<<gblk(128 * 128), blk, 0, stream>>>(mlp_W1, wMlp1t, 128, 128);
  wtrans<<<gblk(128 * 128), blk, 0, stream>>>(mlp_W2, wMlp2t, 128, 128);
  wtrans<<<gblk(128 * 256), blk, 0, stream>>>(fn_W1, wFn1t, 128, 256);
  wtrans<<<gblk(256 * 128), blk, 0, stream>>>(fn_W2, wFn2t, 256, 128);

  // -- time embedding --
  temb_kernel<<<BB, DDD, 0, stream>>>(t, temb_W1, temb_b1, temb_W2, temb_b2, tt);

  // -- edge encoder --
  gather_emb<<<gblk((size_t)EEE * DDD), blk, 0, stream>>>(edge_attr, edge_emb, E0, (size_t)EEE * DDD);
  gemm_wmma<1, 0, 1, 8><<<dim3(2, EEE / 128), blk, 0, stream>>>(E0, wEin1t, ein_b1, nullptr, Etmp, EEE, 128, 256);
  gemm_wmma<0, 1, 0, 8><<<dim3(1, EEE / 128), blk, 0, stream>>>(Etmp, wEin2t, ein_b2, Ebuf, nullptr, EEE, 256, 128);

  // -- densify + time scale: X = scatter_add(e) * tt --
  fzero<<<gblk(XEL), blk, 0, stream>>>(X, XEL);
  scatter_edges<<<gblk((size_t)EEE * DDD), blk, 0, stream>>>(edge_index, Ebuf, X);
  scale_time<<<gblk(XEL), blk, 0, stream>>>(X, tt);

  // -- A = initial X: build per-(b,d) bf16 matrices + transposes (once) --
  make_xm<<<gblk(XEL), blk, 0, stream>>>(X, Am, AmT, nullptr, 0);

  // -- layer loop --
  for (int l = 0; l < LLL; ++l) {
    make_xm<<<gblk(XEL), blk, 0, stream>>>(X, Xm, XmT, Y, 1);          // Y[:,0:D] = X
    einsum_wmma<<<BB * DDD, blk, 0, stream>>>(Am, AmT, Xm, XmT, Y);    // Y[:,D:3D] = X1,X2
    gemm_wmma<0, 1, 0, 8><<<dim3(1, ROWS / 128), blk, 0, stream>>>(
        Y, wConv1t + (size_t)l * 128 * 384, conv_b1 + (size_t)l * 128, T1, nullptr, ROWS, 384, 128);
    ln_relu<0><<<ROWS / 8, blk, 0, stream>>>(T1, conv_g1 + (size_t)l * 128,
                                             conv_bt1 + (size_t)l * 128, T1b, nullptr, ROWS);
    gemm_wmma<0, 1, 0, 8><<<dim3(1, ROWS / 128), blk, 0, stream>>>(
        T1b, wConv2t + (size_t)l * 128 * 128, conv_b2 + (size_t)l * 128, T2, nullptr, ROWS, 128, 128);
    ln_relu<1><<<ROWS / 8, blk, 0, stream>>>(T2, conv_g2 + (size_t)l * 128,
                                             conv_bt2 + (size_t)l * 128, nullptr, X, ROWS);  // X += relu(ln)
  }

  // -- X = X + X^T; gather edge features --
  symmetrize<<<gblk(XEL), blk, 0, stream>>>(X, XsB);
  gather_eout<<<gblk((size_t)EEE * DDD), blk, 0, stream>>>(edge_index, XsB, EoutB);

  // -- node head: mlp over (B,N,N,D), mean over j --
  gemm_wmma<2, 0, 1, 8><<<dim3(1, ROWS / 128), blk, 0, stream>>>(XsB, wMlp1t, mlp_b1, nullptr, M1b, ROWS, 128, 128);
  gemm_wmma<0, 1, 0, 8><<<dim3(1, ROWS / 128), blk, 0, stream>>>(M1b, wMlp2t, mlp_b2, M2, nullptr, ROWS, 128, 128);
  mean_rows64<<<2048, DDD, 0, stream>>>(M2, h2b);

  // -- final layers --
  gemm_wmma<1, 0, 1, 8><<<dim3(2, 2048 / 128), blk, 0, stream>>>(h2b, wFn1t, fn_b1, nullptr, F1, 2048, 128, 256);
  gemm_wmma<0, 1, 0, 8><<<dim3(1, 2048 / 128), blk, 0, stream>>>(F1, wFn2t, fn_b2, node_out, nullptr, 2048, 256, 128);
  gemm_wmma<1, 0, 1, 8><<<dim3(2, EEE / 128), blk, 0, stream>>>(EoutB, wFn1t, fn_b1, nullptr, F1, EEE, 128, 256);
  gemm_wmma<0, 1, 0, 8><<<dim3(1, EEE / 128), blk, 0, stream>>>(F1, wFn2t, fn_b2, edge_out, nullptr, EEE, 256, 128);

  graph_mean<<<BB, DDD, 0, stream>>>(node_out, gattr);
}
